// Attention_47725676593424
// MI455X (gfx1250) — compile-verified, matrix operand order checked by reading, not run on hardware
//
#include <hip/hip_runtime.h>
#include <hip/hip_bf16.h>

// ---------------------------------------------------------------------------
// Problem constants (match reference)
// ---------------------------------------------------------------------------
#define B_    2
#define S_    2048
#define D_    1024
#define H_    16
#define KVH_  4
#define HD_   64
#define BS_   (B_ * S_)

typedef __bf16 bf16_t;
typedef __attribute__((ext_vector_type(16))) __bf16 v16bf;
typedef __attribute__((ext_vector_type(8)))  __bf16 v8bf;   // 16 bytes
typedef __attribute__((ext_vector_type(8)))  float  v8f;
typedef __attribute__((ext_vector_type(4)))  unsigned int u32x4;
typedef __attribute__((ext_vector_type(8)))  unsigned int u32x8;

// A v16bf WMMA fragment is two contiguous 16-byte chunks per lane:
//   elements 0..7  = K (klo .. klo+7)        -> chunk at base + klo
//   elements 8..15 = K (16+klo .. 16+klo+7)  -> chunk at base + 16 + klo
// where klo = 8*(lane>=16).  (CDNA5 ISA 7.12.2, 16-bit A/B 16x32 layout.)
union frag_t { v16bf v; v8bf h[2]; };

__device__ __forceinline__ v8f wmma_bf16(const v16bf& a, const v16bf& b, const v8f& c) {
    return __builtin_amdgcn_wmma_f32_16x16x32_bf16(false, a, false, b, (short)0, c,
                                                   false, false);
}

// ---------------------------------------------------------------------------
// TDM: issue TENSOR_LOAD_TO_LDS of a rows x 32-element bf16 tile whose rows
// are K elements apart, from tile start gsrc into LDS byte offset lds_off.
// D# per CDNA5 ISA 8.3/8.4 (2 SGPR groups; VADDR2/3 omitted -> NULL).
// Wave-level op: EXEC ignored; tracked by TENSORcnt.
// ---------------------------------------------------------------------------
__device__ __forceinline__ void tdm_load_tile(const bf16_t* gsrc, unsigned lds_off,
                                              unsigned K, unsigned M, unsigned rows) {
    const unsigned long long ga = (unsigned long long)(uintptr_t)gsrc;
    u32x4 g0;
    g0[0] = 1u;                                         // count=1, user mode
    g0[1] = lds_off;                                    // lds_addr
    g0[2] = (unsigned)(ga & 0xFFFFFFFFu);               // global_addr[31:0]
    g0[3] = (unsigned)((ga >> 32) & 0x01FFFFFFu)        // global_addr[56:32]
          | (2u << 30);                                 // type = 2 ("image")
    u32x8 g1;
    g1[0] = 0x00010000u;                                // data_size=1 (2 bytes)
    g1[1] = (K & 0xFFFFu) << 16;                        // tensor_dim0[15:0]
    g1[2] = ((K >> 16) & 0xFFFFu)                       // tensor_dim0[31:16]
          | ((M & 0xFFFFu) << 16);                      // tensor_dim1[15:0]
    g1[3] = ((M >> 16) & 0xFFFFu)                       // tensor_dim1[31:16]
          | (32u << 16);                                // tile_dim0 = 32
    g1[4] = rows & 0xFFFFu;                             // tile_dim1 = rows, tile_dim2=0
    g1[5] = K;                                          // tensor_dim0_stride[31:0]
    g1[6] = 0u;                                         // stride0 hi, stride1 lo
    g1[7] = 0u;                                         // stride1 hi
    asm volatile("tensor_load_to_lds %0, %1" :: "s"(g0), "s"(g1) : "memory");
}

// ---------------------------------------------------------------------------
// f32 -> bf16 elementwise convert
// ---------------------------------------------------------------------------
__global__ void __launch_bounds__(256)
k_f32_to_bf16(const float* __restrict__ in, bf16_t* __restrict__ out, int n) {
    int i = blockIdx.x * blockDim.x + threadIdx.x;
    if (i < n) out[i] = (bf16_t)in[i];
}

// ---------------------------------------------------------------------------
// f32 [K,N] -> bf16 transposed [N,K] (weights pre-transposed so WMMA B
// fragments are contiguous 16B chunks in global memory)
// ---------------------------------------------------------------------------
__global__ void __launch_bounds__(256)
k_transpose_bf16(const float* __restrict__ in, bf16_t* __restrict__ out,
                 int K, int N) {
    int idx = blockIdx.x * blockDim.x + threadIdx.x;
    if (idx >= K * N) return;
    const int k = idx / N;
    const int n = idx - k * N;
    out[(size_t)n * K + k] = (bf16_t)in[idx];
}

// ---------------------------------------------------------------------------
// bf16 WMMA GEMM: C[M,N](f32) = A[M,K](bf16) x B^T[N,K](bf16, pre-transposed)
// Block: 256 threads = 8 waves, 128x64 C tile.
// A panels (128x32) are DMA'd into LDS by the Tensor Data Mover,
// double-buffered so the TDM overlaps WMMA compute. B fragments load
// straight from global W^T as 16B chunks (panels live in L2).
// Requires M%128==0, N%64==0, K%32==0.
// ---------------------------------------------------------------------------
#define GM_TM 128
#define GM_TN 64
__global__ void __launch_bounds__(256)
k_gemm_bf16(const bf16_t* __restrict__ A, const bf16_t* __restrict__ BwT,
            float* __restrict__ C, int M, int N, int K) {
    __shared__ __align__(16) bf16_t Asm[2][GM_TM][32];   // 16 KB, double buffer

    const int tid  = threadIdx.x;
    const int lane = tid & 31;
    const int wv   = tid >> 5;
    const int nbn  = N / GM_TN;
    const int bm   = blockIdx.x / nbn;
    const int bn   = blockIdx.x - bm * nbn;
    const int m0   = bm * GM_TM;
    const int n0   = bn * GM_TN;

    const int col  = lane & 15;
    const int half = lane >> 4;
    const int klo  = half << 3;

    v8f acc[4];
#pragma unroll
    for (int g = 0; g < 4; ++g)
#pragma unroll
        for (int r = 0; r < 8; ++r) acc[g][r] = 0.f;

    const unsigned lds0 = (unsigned)(uintptr_t)(&Asm[0][0][0]);
    const unsigned lds1 = (unsigned)(uintptr_t)(&Asm[1][0][0]);
    const int nk = K >> 5;

    // prime the pipeline: TDM panel 0 -> buffer 0
    if (wv == 0) tdm_load_tile(A + (size_t)m0 * K, lds0, K, M, GM_TM);

    for (int i = 0; i < nk; ++i) {
        const int kk = i << 5;
        if (wv == 0) {
            if (i + 1 < nk) {
                // issue next panel into the other buffer, then wait until the
                // current one (older, in-order) has landed: TENSORcnt <= 1
                tdm_load_tile(A + (size_t)m0 * K + (kk + 32),
                              (i & 1) ? lds0 : lds1, K, M, GM_TM);
                __builtin_amdgcn_s_wait_tensorcnt(1);
            } else {
                __builtin_amdgcn_s_wait_tensorcnt(0);
            }
        }
        __syncthreads();                        // panel i visible to all waves

        // ---- compute: 1 A fragment reused across 4 WMMAs ----
        frag_t af;
        const int arow = (wv << 4) + col;
        const bf16_t* Ar = &Asm[i & 1][arow][0];
        af.h[0] = *(const v8bf*)(Ar + klo);
        af.h[1] = *(const v8bf*)(Ar + 16 + klo);
#pragma unroll
        for (int g = 0; g < 4; ++g) {
            const bf16_t* Bn = BwT + (size_t)(n0 + (g << 4) + col) * K + kk;
            frag_t bfr;
            bfr.h[0] = *(const v8bf*)(Bn + klo);
            bfr.h[1] = *(const v8bf*)(Bn + 16 + klo);
            acc[g] = wmma_bf16(af.v, bfr.v, acc[g]);
        }
        if (i + 1 < nk)                         // prefetch next W^T panel
            __builtin_prefetch(BwT + (size_t)(n0 + (tid >> 2)) * K + kk + 32, 0, 3);
        __syncthreads();                        // protect buffer reuse
    }

    // ---- epilogue ----
    float* Cw = C + (size_t)(m0 + (wv << 4)) * N + n0;
#pragma unroll
    for (int g = 0; g < 4; ++g)
#pragma unroll
        for (int r = 0; r < 8; ++r) {
            const int m = r + (half << 3);
            Cw[(size_t)m * N + (g << 4) + col] = acc[g][r];
        }
}

// ---------------------------------------------------------------------------
// RoPE on Q: q_raw [BS, H*HD] f32 -> q_bf [b,h,s,hd] bf16
// ---------------------------------------------------------------------------
__global__ void __launch_bounds__(256)
k_rope_q(const float* __restrict__ q_raw, const float* __restrict__ cosT,
         const float* __restrict__ sinT, bf16_t* __restrict__ q_bf) {
    int idx = blockIdx.x * blockDim.x + threadIdx.x;     // B*S*H*HD = 2^22
    if (idx >= B_ * S_ * H_ * HD_) return;
    const int hd = idx & 63;
    const int h  = (idx >> 6) & 15;
    const int s  = (idx >> 10) & (S_ - 1);
    const int b  = idx >> 21;
    const float* qr = q_raw + (size_t)(b * S_ + s) * (H_ * HD_) + h * HD_;
    float v;
    if (hd < 32) {
        v = qr[hd] * cosT[s * 32 + hd] - qr[hd + 32] * sinT[s * 32 + hd];
    } else {
        v = qr[hd] * cosT[s * 32 + hd - 32] + qr[hd - 32] * sinT[s * 32 + hd - 32];
    }
    q_bf[((size_t)(b * H_ + h) * S_ + s) * HD_ + hd] = (bf16_t)v;
}

// ---------------------------------------------------------------------------
// RoPE on K: k_raw [BS, KVH*HD] -> k_bf [b,kh,s,hd] bf16 + new_k f32
// ---------------------------------------------------------------------------
__global__ void __launch_bounds__(256)
k_rope_k(const float* __restrict__ k_raw, const float* __restrict__ cosT,
         const float* __restrict__ sinT, bf16_t* __restrict__ k_bf,
         float* __restrict__ new_k) {
    int idx = blockIdx.x * blockDim.x + threadIdx.x;     // B*S*KVH*HD = 2^20
    if (idx >= B_ * S_ * KVH_ * HD_) return;
    const int hd = idx & 63;
    const int kh = (idx >> 6) & 3;
    const int s  = (idx >> 8) & (S_ - 1);
    const int b  = idx >> 19;
    const float* kr = k_raw + (size_t)(b * S_ + s) * (KVH_ * HD_) + kh * HD_;
    float v;
    if (hd < 32) {
        v = kr[hd] * cosT[s * 32 + hd] - kr[hd + 32] * sinT[s * 32 + hd];
    } else {
        v = kr[hd] * cosT[s * 32 + hd - 32] + kr[hd - 32] * sinT[s * 32 + hd - 32];
    }
    k_bf[((size_t)(b * KVH_ + kh) * S_ + s) * HD_ + hd] = (bf16_t)v;
    new_k[((size_t)(b * S_ + s) * KVH_ + kh) * HD_ + hd] = v;
}

// ---------------------------------------------------------------------------
// V: v_raw -> v_t bf16 TRANSPOSED [b,kh,hd,s] + new_v f32
// ---------------------------------------------------------------------------
__global__ void __launch_bounds__(256)
k_copy_v(const float* __restrict__ v_raw, bf16_t* __restrict__ v_t,
         float* __restrict__ new_v) {
    int idx = blockIdx.x * blockDim.x + threadIdx.x;
    if (idx >= B_ * S_ * KVH_ * HD_) return;
    const int hd = idx & 63;
    const int kh = (idx >> 6) & 3;
    const int s  = (idx >> 8) & (S_ - 1);
    const int b  = idx >> 19;
    const float v = v_raw[(size_t)(b * S_ + s) * (KVH_ * HD_) + kh * HD_ + hd];
    v_t[((size_t)(b * KVH_ + kh) * HD_ + hd) * S_ + s] = (bf16_t)v;
    new_v[((size_t)(b * S_ + s) * KVH_ + kh) * HD_ + hd] = v;
}

// ---------------------------------------------------------------------------
// Flash attention (causal, GQA). One wave = one (b, h, 16-query tile).
// 32 keys/iter: 4 score WMMAs + 4 full-K P.V WMMAs; all fragment loads 16B.
// ---------------------------------------------------------------------------
#define FA_WAVES 8
__global__ void __launch_bounds__(FA_WAVES * 32)
k_flash_attn(const bf16_t* __restrict__ q_bf, const bf16_t* __restrict__ k_bf,
             const bf16_t* __restrict__ v_t, bf16_t* __restrict__ o_bf) {
    __shared__ __align__(16) bf16_t pshare[FA_WAVES][16 * 32];   // 8 KB

    const int wib  = threadIdx.x >> 5;
    const int lane = threadIdx.x & 31;
    const int gw   = blockIdx.x * FA_WAVES + wib;   // 0..4095
    const int qt   = gw & 127;                      // S/16 = 128
    const int h    = (gw >> 7) & 15;
    const int b    = gw >> 11;
    const int kh   = h >> 2;                        // GQA n_rep = 4

    const int col  = lane & 15;
    const int half = lane >> 4;
    const int klo  = half << 3;

    const bf16_t* Qrow = q_bf + ((size_t)(b * H_ + h) * S_ + qt * 16 + col) * HD_;
    frag_t aq0, aq1;
    aq0.h[0] = *(const v8bf*)(Qrow + klo);
    aq0.h[1] = *(const v8bf*)(Qrow + 16 + klo);
    aq1.h[0] = *(const v8bf*)(Qrow + 32 + klo);
    aq1.h[1] = *(const v8bf*)(Qrow + 48 + klo);

    v8f o[4];
#pragma unroll
    for (int g = 0; g < 4; ++g)
#pragma unroll
        for (int r = 0; r < 8; ++r) o[g][r] = 0.f;

    float mrun[8], lrun[8];
#pragma unroll
    for (int r = 0; r < 8; ++r) { mrun[r] = -1e30f; lrun[r] = 0.f; }

    const bf16_t* Kh = k_bf + (size_t)(b * KVH_ + kh) * S_ * HD_;
    const bf16_t* Vh = v_t  + (size_t)(b * KVH_ + kh) * HD_ * S_;
    bf16_t* pl = &pshare[wib][0];

    const int nb = (qt + 2) >> 1;                   // 32-key blocks, <= 64
    for (int kb = 0; kb < nb; ++kb) {
        const int key0 = kb << 5;

        v8f sc0, sc1;
#pragma unroll
        for (int g2 = 0; g2 < 2; ++g2) {
            const bf16_t* Krow = Kh + (size_t)(key0 + (g2 << 4) + col) * HD_;
            frag_t bkA, bkB;
            bkA.h[0] = *(const v8bf*)(Krow + klo);
            bkA.h[1] = *(const v8bf*)(Krow + 16 + klo);
            bkB.h[0] = *(const v8bf*)(Krow + 32 + klo);
            bkB.h[1] = *(const v8bf*)(Krow + 48 + klo);
            v8f s;
#pragma unroll
            for (int r = 0; r < 8; ++r) s[r] = 0.f;
            s = wmma_bf16(aq0.v, bkA.v, s);
            s = wmma_bf16(aq1.v, bkB.v, s);
            if (g2 == 0) sc0 = s; else sc1 = s;
        }

        const bool last = (kb == nb - 1);
#pragma unroll
        for (int r = 0; r < 8; ++r) {
            const int grow = (qt << 4) + r + (half << 3);
            float x0 = sc0[r] * 0.125f;             // 1/sqrt(64)
            float x1 = sc1[r] * 0.125f;
            if (last) {
                if (key0 + col > grow)      x0 = -1e30f;
                if (key0 + 16 + col > grow) x1 = -1e30f;
            }
            sc0[r] = x0; sc1[r] = x1;
        }

#pragma unroll
        for (int r = 0; r < 8; ++r) {
            float rm = fmaxf(sc0[r], sc1[r]);
            rm = fmaxf(rm, __shfl_xor(rm, 1, 32));
            rm = fmaxf(rm, __shfl_xor(rm, 2, 32));
            rm = fmaxf(rm, __shfl_xor(rm, 4, 32));
            rm = fmaxf(rm, __shfl_xor(rm, 8, 32));
            const float nm   = fmaxf(mrun[r], rm);
            const float p0   = __expf(sc0[r] - nm);
            const float p1   = __expf(sc1[r] - nm);
            const float corr = __expf(mrun[r] - nm);
            float rs = p0 + p1;
            rs += __shfl_xor(rs, 1, 32);
            rs += __shfl_xor(rs, 2, 32);
            rs += __shfl_xor(rs, 4, 32);
            rs += __shfl_xor(rs, 8, 32);
            lrun[r] = lrun[r] * corr + rs;
            mrun[r] = nm;
            o[0][r] *= corr; o[1][r] *= corr; o[2][r] *= corr; o[3][r] *= corr;
            const int m = r + (half << 3);
            pl[(m << 5) + col]      = (bf16_t)p0;
            pl[(m << 5) + 16 + col] = (bf16_t)p1;
        }

        frag_t ap;
        const int prow = (lane & 15) << 5;
        ap.h[0] = *(const v8bf*)(pl + prow + klo);
        ap.h[1] = *(const v8bf*)(pl + prow + 16 + klo);

#pragma unroll
        for (int g = 0; g < 4; ++g) {
            const bf16_t* VTrow = Vh + (size_t)((g << 4) + col) * S_ + key0;
            frag_t bv;
            bv.h[0] = *(const v8bf*)(VTrow + klo);
            bv.h[1] = *(const v8bf*)(VTrow + 16 + klo);
            o[g] = wmma_bf16(ap.v, bv.v, o[g]);
        }
    }

#pragma unroll
    for (int r = 0; r < 8; ++r) {
        const float inv = 1.f / lrun[r];
        const int   m   = r + (half << 3);
        const size_t ro = (size_t)(b * S_ + (qt << 4) + m) * (H_ * HD_) + h * HD_;
        o_bf[ro + 0 * 16 + col] = (bf16_t)(o[0][r] * inv);
        o_bf[ro + 1 * 16 + col] = (bf16_t)(o[1][r] * inv);
        o_bf[ro + 2 * 16 + col] = (bf16_t)(o[2][r] * inv);
        o_bf[ro + 3 * 16 + col] = (bf16_t)(o[3][r] * inv);
    }
}

// ---------------------------------------------------------------------------
// Host-side orchestration
// ---------------------------------------------------------------------------
extern "C" void kernel_launch(void* const* d_in, const int* in_sizes, int n_in,
                              void* d_out, int out_size, void* d_ws, size_t ws_size,
                              hipStream_t stream) {
    const float* x    = (const float*)d_in[0];
    const float* cosT = (const float*)d_in[1];
    const float* sinT = (const float*)d_in[2];
    // d_in[3] = mask: causal mask applied analytically in-kernel
    const float* wq   = (const float*)d_in[4];
    const float* wk   = (const float*)d_in[5];
    const float* wv   = (const float*)d_in[6];
    const float* wo   = (const float*)d_in[7];

    float* out   = (float*)d_out;                       // [BS, D]
    float* new_k = out + (size_t)BS_ * D_;              // [B,S,KVH,HD]
    float* new_v = new_k + (size_t)BS_ * KVH_ * HD_;    // [B,S,KVH,HD]

    // ---- workspace carve-up ----
    char* ws = (char*)d_ws;
    size_t off = 0;
    auto carve = [&](size_t bytes) {
        char* p = ws + off; off += (bytes + 255) & ~(size_t)255; return p;
    };

    bf16_t* x_bf  = (bf16_t*)carve((size_t)BS_ * D_ * 2);
    bf16_t* wqT   = (bf16_t*)carve((size_t)D_ * H_ * HD_ * 2);     // [1024][1024]
    bf16_t* wkT   = (bf16_t*)carve((size_t)D_ * KVH_ * HD_ * 2);   // [256][1024]
    bf16_t* wvT   = (bf16_t*)carve((size_t)D_ * KVH_ * HD_ * 2);   // [256][1024]
    bf16_t* woT   = (bf16_t*)carve((size_t)(H_ * HD_) * D_ * 2);   // [1024][1024]
    float*  q_raw = (float*)carve((size_t)BS_ * H_ * HD_ * 4);
    float*  k_raw = (float*)carve((size_t)BS_ * KVH_ * HD_ * 4);
    float*  v_raw = (float*)carve((size_t)BS_ * KVH_ * HD_ * 4);
    bf16_t* q_bf  = (bf16_t*)carve((size_t)BS_ * H_ * HD_ * 2);
    bf16_t* k_bf  = (bf16_t*)carve((size_t)BS_ * KVH_ * HD_ * 2);
    bf16_t* v_t   = (bf16_t*)carve((size_t)BS_ * KVH_ * HD_ * 2);
    bf16_t* attn_bf = (bf16_t*)q_raw;   // alias: q_raw dead after RoPE

    const int thr = 256;

    // ---- 1) x -> bf16 ; weights -> transposed bf16 ----
    k_f32_to_bf16<<<(BS_ * D_) / thr, thr, 0, stream>>>(x, x_bf, BS_ * D_);
    k_transpose_bf16<<<(D_ * H_ * HD_) / thr, thr, 0, stream>>>(wq, wqT, D_, H_ * HD_);
    k_transpose_bf16<<<(D_ * KVH_ * HD_) / thr, thr, 0, stream>>>(wk, wkT, D_, KVH_ * HD_);
    k_transpose_bf16<<<(D_ * KVH_ * HD_) / thr, thr, 0, stream>>>(wv, wvT, D_, KVH_ * HD_);
    k_transpose_bf16<<<(H_ * HD_ * D_) / thr, thr, 0, stream>>>(wo, woT, H_ * HD_, D_);

    // ---- 2) QKV projections (TDM-fed WMMA GEMMs) ----
    k_gemm_bf16<<<(BS_ / GM_TM) * ((H_ * HD_) / GM_TN), thr, 0, stream>>>(
        x_bf, wqT, q_raw, BS_, H_ * HD_, D_);
    k_gemm_bf16<<<(BS_ / GM_TM) * ((KVH_ * HD_) / GM_TN), thr, 0, stream>>>(
        x_bf, wkT, k_raw, BS_, KVH_ * HD_, D_);
    k_gemm_bf16<<<(BS_ / GM_TM) * ((KVH_ * HD_) / GM_TN), thr, 0, stream>>>(
        x_bf, wvT, v_raw, BS_, KVH_ * HD_, D_);

    // ---- 3) RoPE + layout change + kv-cache outputs ----
    k_rope_q<<<(B_ * S_ * H_ * HD_) / thr, thr, 0, stream>>>(q_raw, cosT, sinT, q_bf);
    k_rope_k<<<(B_ * S_ * KVH_ * HD_) / thr, thr, 0, stream>>>(k_raw, cosT, sinT, k_bf, new_k);
    k_copy_v<<<(B_ * S_ * KVH_ * HD_) / thr, thr, 0, stream>>>(v_raw, v_t, new_v);

    // ---- 4) causal flash attention (WMMA) ----
    k_flash_attn<<<(B_ * H_ * (S_ / 16)) / FA_WAVES, FA_WAVES * 32, 0, stream>>>(
        q_bf, k_bf, v_t, attn_bf);

    // ---- 5) output projection (TDM-fed WMMA GEMM) -> d_out ----
    k_gemm_bf16<<<(BS_ / GM_TM) * (D_ / GM_TN), thr, 0, stream>>>(
        attn_bf, woT, out, BS_, D_, H_ * HD_);

    (void)in_sizes; (void)n_in; (void)out_size; (void)ws_size;
}
